// FGAEmbedder_22548578304196
// MI455X (gfx1250) — matrix operand order repaired and checked by reading, not compiled, not run on hardware
//
#include <hip/hip_runtime.h>
#include <hip/hip_bf16.h>
#include <math.h>

// ---------------- problem constants ----------------
constexpr int BB   = 32;
constexpr int TT   = 1024;
constexpr int K_IN = 1536;
constexpr int DD   = 768;
constexpr int BT   = BB * TT;   // 32768 rows

typedef __attribute__((ext_vector_type(16))) __bf16 v16bf;
typedef __attribute__((ext_vector_type(8)))  float  v8f;
typedef __attribute__((ext_vector_type(4)))  float  f32x4;

struct __align__(8) Bf16x4 { __bf16 x, y, z, w; };

// ---------------- f32 -> bf16 convert (vectorized) ----------------
__global__ void cvt_f32_bf16(const float* __restrict__ s, __bf16* __restrict__ d, int n4) {
    int i = blockIdx.x * blockDim.x + threadIdx.x;
    if (i >= n4) return;
    f32x4 v = ((const f32x4*)s)[i];
    Bf16x4 o;
    o.x = (__bf16)v.x; o.y = (__bf16)v.y; o.z = (__bf16)v.z; o.w = (__bf16)v.w;
    ((Bf16x4*)d)[i] = o;
}

// ---------------- WMMA GEMM: C = A[M,K] * W[N,K]^T + bias ----------------
// Block: 256 threads (8 waves, 4 in M x 2 in N). Block tile 128(M) x 128(N).
// Each wave: 32x64 sub-tile = 8 accumulators; per K-slice of 32:
//   2 A frags + 4 B frags (12 ds_load_b128) -> 8 WMMAs.
// Double-buffered LDS filled by the CDNA5 async global->LDS engine (ASYNCcnt).
// Main loop manually unrolled x2 so buffer parity is static (nk is even).
// MODE 0: outb = relu(C)                         (fc1 -> h bf16)
// MODE 1: outf = C, outb = bf16(C)               (fc2 -> u f32 + ub bf16)
// MODE 2: rowacc[m] += sum_n relu(C)*redw[n]     (unary potential, atomic)
// MODE 3: outf = C, rowacc[m] += sum_n C*C       (pairwise embed + row sumsq)
template<int MODE>
__global__ __launch_bounds__(256)
void gemm_wmma(const __bf16* __restrict__ A, const __bf16* __restrict__ W,
               const float* __restrict__ bias, int M, int N, int K,
               float* __restrict__ outf, __bf16* __restrict__ outb,
               const float* __restrict__ redw, float* __restrict__ rowacc)
{
    __shared__ __align__(16) __bf16 lA[2][128][40];  // pad 32->40 vs LDS bank conflicts
    __shared__ __align__(16) __bf16 lB[2][128][40];
    constexpr unsigned BUFSTEP = 128 * 40 * 2;       // bytes per buffer

    const int tid  = threadIdx.x;
    const int wv   = tid >> 5;
    const int lane = tid & 31;
    const int n0   = blockIdx.x * 128;
    const int m0   = blockIdx.y * 128;

    v8f acc[2][4];
#pragma unroll
    for (int i = 0; i < 2; ++i)
#pragma unroll
        for (int j = 0; j < 4; ++j)
#pragma unroll
            for (int e = 0; e < 8; ++e) acc[i][j][e] = 0.f;

    // ---- async staging: each thread owns two 16B chunks of A and of W ----
    const int r0 = tid >> 2;            // 0..63 (rows r0 and r0+64)
    const int kc = (tid & 3) * 8;       // K chunk within 32-slice
    const unsigned aoff0 = (unsigned)(((size_t)(m0 + r0)      * K + kc) * 2);
    const unsigned aoff1 = (unsigned)(((size_t)(m0 + r0 + 64) * K + kc) * 2);
    const unsigned boff0 = (unsigned)(((size_t)(n0 + r0)      * K + kc) * 2);
    const unsigned boff1 = (unsigned)(((size_t)(n0 + r0 + 64) * K + kc) * 2);
    const unsigned ldsA0 = (unsigned)(uintptr_t)&lA[0][r0][kc];
    const unsigned ldsA1 = (unsigned)(uintptr_t)&lA[0][r0 + 64][kc];
    const unsigned ldsB0 = (unsigned)(uintptr_t)&lB[0][r0][kc];
    const unsigned ldsB1 = (unsigned)(uintptr_t)&lB[0][r0 + 64][kc];

    auto stage = [&](unsigned sel, unsigned kbytes) {
        asm volatile("global_load_async_to_lds_b128 %0, %1, %2"
                     :: "v"(ldsA0 + sel), "v"(aoff0 + kbytes), "s"(A) : "memory");
        asm volatile("global_load_async_to_lds_b128 %0, %1, %2"
                     :: "v"(ldsA1 + sel), "v"(aoff1 + kbytes), "s"(A) : "memory");
        asm volatile("global_load_async_to_lds_b128 %0, %1, %2"
                     :: "v"(ldsB0 + sel), "v"(boff0 + kbytes), "s"(W) : "memory");
        asm volatile("global_load_async_to_lds_b128 %0, %1, %2"
                     :: "v"(ldsB1 + sel), "v"(boff1 + kbytes), "s"(W) : "memory");
    };

    // fragment addressing (contiguous k order, same in A and B)
    const int wvM  = wv & 3;                  // 0..3 : 32-row strip
    const int wvN  = wv >> 2;                 // 0..1 : 64-col strip
    const int arow = wvM * 32 + (lane & 15);  // A rows arow / arow+16
    const int ka   = (lane >> 4) * 16;        // A frag: K 0..15 / 16..31 contiguous
    const int kb   = (lane >> 4) * 16;        // B frag: K 0..15 / 16..31 contiguous
    const int bc   = wvN * 64 + (lane & 15);  // B col base within block tile

    auto compute = [&](const __bf16 (*cA)[40], const __bf16 (*cB)[40]) {
        union V16 { v16bf v; f32x4 f[2]; } a0, a1, b0, b1, b2, b3;
        a0.f[0] = *(const f32x4*)&cA[arow][ka];
        a0.f[1] = *(const f32x4*)&cA[arow][ka + 8];
        a1.f[0] = *(const f32x4*)&cA[arow + 16][ka];
        a1.f[1] = *(const f32x4*)&cA[arow + 16][ka + 8];
        b0.f[0] = *(const f32x4*)&cB[bc][kb];
        b0.f[1] = *(const f32x4*)&cB[bc][kb + 8];
        b1.f[0] = *(const f32x4*)&cB[16 + bc][kb];
        b1.f[1] = *(const f32x4*)&cB[16 + bc][kb + 8];
        b2.f[0] = *(const f32x4*)&cB[32 + bc][kb];
        b2.f[1] = *(const f32x4*)&cB[32 + bc][kb + 8];
        b3.f[0] = *(const f32x4*)&cB[48 + bc][kb];
        b3.f[1] = *(const f32x4*)&cB[48 + bc][kb + 8];

        acc[0][0] = __builtin_amdgcn_wmma_f32_16x16x32_bf16(false, a0.v, false, b0.v, (short)0, acc[0][0], false, false);
        acc[0][1] = __builtin_amdgcn_wmma_f32_16x16x32_bf16(false, a0.v, false, b1.v, (short)0, acc[0][1], false, false);
        acc[0][2] = __builtin_amdgcn_wmma_f32_16x16x32_bf16(false, a0.v, false, b2.v, (short)0, acc[0][2], false, false);
        acc[0][3] = __builtin_amdgcn_wmma_f32_16x16x32_bf16(false, a0.v, false, b3.v, (short)0, acc[0][3], false, false);
        acc[1][0] = __builtin_amdgcn_wmma_f32_16x16x32_bf16(false, a1.v, false, b0.v, (short)0, acc[1][0], false, false);
        acc[1][1] = __builtin_amdgcn_wmma_f32_16x16x32_bf16(false, a1.v, false, b1.v, (short)0, acc[1][1], false, false);
        acc[1][2] = __builtin_amdgcn_wmma_f32_16x16x32_bf16(false, a1.v, false, b2.v, (short)0, acc[1][2], false, false);
        acc[1][3] = __builtin_amdgcn_wmma_f32_16x16x32_bf16(false, a1.v, false, b3.v, (short)0, acc[1][3], false, false);
    };

    const int nk = K >> 5;                     // always even here (48 or 24)
    stage(0u, 0u);

    int ks = 0;
    for (; ks < nk - 2; ks += 2) {
        asm volatile("s_wait_asynccnt 0x0" ::: "memory");
        __syncthreads();
        stage(BUFSTEP, (unsigned)(ks + 1) * 64u);
        compute(lA[0], lB[0]);

        asm volatile("s_wait_asynccnt 0x0" ::: "memory");
        __syncthreads();
        stage(0u, (unsigned)(ks + 2) * 64u);
        compute(lA[1], lB[1]);
    }
    // tail: slices nk-2 (buf0) and nk-1 (buf1)
    asm volatile("s_wait_asynccnt 0x0" ::: "memory");
    __syncthreads();
    stage(BUFSTEP, (unsigned)(nk - 1) * 64u);
    compute(lA[0], lB[0]);
    asm volatile("s_wait_asynccnt 0x0" ::: "memory");
    __syncthreads();
    compute(lA[1], lB[1]);

    // epilogue. C layout: VGPR r -> (M = r + 8*(lane>=16), N = lane&15)
    const int mwave = m0 + wvM * 32 + ((lane >> 4) << 3);
    const int nbase = n0 + wvN * 64;
    float bv[4], rv[4];
#pragma unroll
    for (int nt = 0; nt < 4; ++nt) {
        int n = nbase + nt * 16 + (lane & 15);
        bv[nt] = bias[n];
        rv[nt] = (MODE == 2) ? redw[n] : 0.f;
    }

    if (MODE == 0 || MODE == 1) {
#pragma unroll
        for (int mi = 0; mi < 2; ++mi)
#pragma unroll
            for (int nt = 0; nt < 4; ++nt) {
                int n = nbase + nt * 16 + (lane & 15);
#pragma unroll
                for (int r = 0; r < 8; ++r) {
                    float v = acc[mi][nt][r] + bv[nt];
                    if (MODE == 0) v = fmaxf(v, 0.f);
                    size_t idx = (size_t)(mwave + mi * 16 + r) * N + n;
                    if (MODE == 1) outf[idx] = v;
                    outb[idx] = (__bf16)v;
                }
            }
    } else {
#pragma unroll
        for (int mi = 0; mi < 2; ++mi) {
            float rs[8];
#pragma unroll
            for (int r = 0; r < 8; ++r) rs[r] = 0.f;
#pragma unroll
            for (int nt = 0; nt < 4; ++nt) {
                int n = nbase + nt * 16 + (lane & 15);
#pragma unroll
                for (int r = 0; r < 8; ++r) {
                    float v = acc[mi][nt][r] + bv[nt];
                    if (MODE == 2) { v = fmaxf(v, 0.f); rs[r] += v * rv[nt]; }
                    else {
                        size_t idx = (size_t)(mwave + mi * 16 + r) * N + n;
                        outf[idx] = v;
                        rs[r] += v * v;
                    }
                }
            }
            // reduce across the 16 lanes sharing a row (xor 1,2,4,8 stays in half)
#pragma unroll
            for (int r = 0; r < 8; ++r) {
                float s = rs[r];
                s += __shfl_xor(s, 1, 32);
                s += __shfl_xor(s, 2, 32);
                s += __shfl_xor(s, 4, 32);
                s += __shfl_xor(s, 8, 32);
                if ((lane & 15) == 0) atomicAdd(&rowacc[mwave + mi * 16 + r], s);
            }
        }
    }
}

// ---------------- 1/max(sqrt(ss),1e-12) for both norm arrays ----------------
__global__ void rinv_kernel(const float* __restrict__ xss, const float* __restrict__ yss,
                            float* __restrict__ xinv, float* __restrict__ yinv) {
    int i = blockIdx.x * blockDim.x + threadIdx.x;
    if (i < BT) {
        xinv[i] = 1.f / fmaxf(sqrtf(xss[i]), 1e-12f);
        yinv[i] = 1.f / fmaxf(sqrtf(yss[i]), 1e-12f);
    }
}

// ---------------- ybar[b,d] = (1/T) sum_t ye[b,t,d] * yinv[b,t] ----------------
__global__ __launch_bounds__(256)
void ybar_kernel(const float* __restrict__ ye, const float* __restrict__ yinv,
                 float* __restrict__ ybar) {
    int b = blockIdx.y;
    int d = blockIdx.x * blockDim.x + threadIdx.x;   // < 768
    const float* yp = ye + (size_t)b * TT * DD + d;
    const float* iv = yinv + b * TT;
    float acc = 0.f;
    for (int t = 0; t < TT; ++t) acc += yp[(size_t)t * DD] * iv[t];
    ybar[b * DD + d] = acc * (1.0f / TT);
}

// ---------------- per-batch scores + softmax -> wgt[b,t] ----------------
__global__ __launch_bounds__(256)
void score_softmax_kernel(const float* __restrict__ xe, const float* __restrict__ ybar,
                          const float* __restrict__ xinv, const float* __restrict__ unpot,
                          const float* __restrict__ unr_b, const float* __restrict__ red_w,
                          float* __restrict__ wgt)
{
    __shared__ float sc[TT];
    __shared__ float syb[DD];
    __shared__ float rbuf[256];
    const int b = blockIdx.x;
    const int tid = threadIdx.x, wv = tid >> 5, ln = tid & 31;
    const float rw0 = red_w[0], rw1 = red_w[1], ub = unr_b[0];

    for (int i = tid; i < DD; i += 256) syb[i] = ybar[b * DD + i];
    __syncthreads();

    // pw_pot[t] = (xe[t] . ybar) * xinv[t];  one wave per row
    for (int t = wv; t < TT; t += 8) {
        const float* xr = xe + (size_t)(b * TT + t) * DD;
        float acc = 0.f;
        for (int d = ln; d < DD; d += 32) acc += xr[d] * syb[d];
#pragma unroll
        for (int o = 16; o > 0; o >>= 1) acc += __shfl_xor(acc, o, 32);
        if (ln == 0) {
            int m = b * TT + t;
            sc[t] = rw0 * (unpot[m] + ub) + rw1 * (acc * xinv[m]);
        }
    }
    __syncthreads();

    float mx = -3.4e38f;
    for (int t = tid; t < TT; t += 256) mx = fmaxf(mx, sc[t]);
    rbuf[tid] = mx; __syncthreads();
    for (int s = 128; s > 0; s >>= 1) {
        if (tid < s) rbuf[tid] = fmaxf(rbuf[tid], rbuf[tid + s]);
        __syncthreads();
    }
    mx = rbuf[0]; __syncthreads();

    float sum = 0.f;
    for (int t = tid; t < TT; t += 256) { float e = expf(sc[t] - mx); sc[t] = e; sum += e; }
    rbuf[tid] = sum; __syncthreads();
    for (int s = 128; s > 0; s >>= 1) {
        if (tid < s) rbuf[tid] += rbuf[tid + s];
        __syncthreads();
    }
    float inv = 1.f / rbuf[0];
    for (int t = tid; t < TT; t += 256) wgt[b * TT + t] = sc[t] * inv;
}

// ---------------- out[b,d] = sum_t u[b,t,d] * wgt[b,t] ----------------
__global__ __launch_bounds__(256)
void out_kernel(const float* __restrict__ u, const float* __restrict__ wgt,
                float* __restrict__ out) {
    int b = blockIdx.y;
    int d = blockIdx.x * blockDim.x + threadIdx.x;   // < 768
    const float* up = u + (size_t)b * TT * DD + d;
    const float* wp = wgt + b * TT;
    float acc = 0.f;
    for (int t = 0; t < TT; ++t) acc += up[(size_t)t * DD] * wp[t];
    out[b * DD + d] = acc;
}

// ---------------- host launcher ----------------
extern "C" void kernel_launch(void* const* d_in, const int* in_sizes, int n_in,
                              void* d_out, int out_size, void* d_ws, size_t ws_size,
                              hipStream_t stream)
{
    (void)in_sizes; (void)n_in; (void)out_size; (void)ws_size;
    const float* x     = (const float*)d_in[0];
    const float* fc1w  = (const float*)d_in[1];
    const float* fc1b  = (const float*)d_in[2];
    const float* fc2w  = (const float*)d_in[3];
    const float* fc2b  = (const float*)d_in[4];
    const float* unw   = (const float*)d_in[5];
    const float* unb   = (const float*)d_in[6];
    const float* unrw  = (const float*)d_in[7];
    const float* unrb  = (const float*)d_in[8];
    const float* pxw   = (const float*)d_in[9];
    const float* pxb   = (const float*)d_in[10];
    const float* pyw   = (const float*)d_in[11];
    const float* pyb   = (const float*)d_in[12];
    const float* redw  = (const float*)d_in[13];
    float* out = (float*)d_out;

    char* ws = (char*)d_ws;
    size_t off = 0;
    auto alloc = [&](size_t bytes) {
        size_t r = off; off += (bytes + 255) & ~(size_t)255; return r;
    };
    const size_t o_xb   = alloc((size_t)BT * K_IN * 2); // x bf16;   later aliased: xe f32 (same size)
    const size_t o_hb   = alloc((size_t)BT * K_IN * 2); // h bf16;   later aliased: ye f32 (same size)
    const size_t o_u    = alloc((size_t)BT * DD * 4);   // u f32
    const size_t o_ub   = alloc((size_t)BT * DD * 2);   // u bf16
    const size_t o_wfc1 = alloc((size_t)K_IN * K_IN * 2);
    const size_t o_wfc2 = alloc((size_t)DD * K_IN * 2);
    const size_t o_wun  = alloc((size_t)DD * DD * 2);
    const size_t o_wpx  = alloc((size_t)DD * DD * 2);
    const size_t o_wpy  = alloc((size_t)DD * DD * 2);
    const size_t o_unp  = alloc((size_t)BT * 4);        // unary potential accum
    const size_t o_xss  = alloc((size_t)BT * 4);        // xe row sumsq
    const size_t o_yss  = alloc((size_t)BT * 4);        // ye row sumsq
    const size_t o_xiv  = alloc((size_t)BT * 4);
    const size_t o_yiv  = alloc((size_t)BT * 4);
    const size_t o_ybar = alloc((size_t)BB * DD * 4);
    const size_t o_wgt  = alloc((size_t)BT * 4);

    __bf16* xb   = (__bf16*)(ws + o_xb);
    __bf16* hb   = (__bf16*)(ws + o_hb);
    float*  u    = (float*)(ws + o_u);
    __bf16* ubf  = (__bf16*)(ws + o_ub);
    __bf16* wfc1 = (__bf16*)(ws + o_wfc1);
    __bf16* wfc2 = (__bf16*)(ws + o_wfc2);
    __bf16* wun  = (__bf16*)(ws + o_wun);
    __bf16* wpx  = (__bf16*)(ws + o_wpx);
    __bf16* wpy  = (__bf16*)(ws + o_wpy);
    float*  unp  = (float*)(ws + o_unp);
    float*  xss  = (float*)(ws + o_xss);
    float*  yss  = (float*)(ws + o_yss);
    float*  xiv  = (float*)(ws + o_xiv);
    float*  yiv  = (float*)(ws + o_yiv);
    float*  ybar = (float*)(ws + o_ybar);
    float*  wgt  = (float*)(ws + o_wgt);
    float*  xe   = (float*)(ws + o_xb);   // alias (fc1 staging dead by then)
    float*  ye   = (float*)(ws + o_hb);   // alias (fc2 staging dead by then)

    // zero the atomic accumulators (unp, xss, yss are contiguous)
    hipMemsetAsync(ws + o_unp, 0, o_yss + (size_t)BT * 4 - o_unp, stream);

    // f32 -> bf16 conversions
    auto cvt = [&](const float* s, __bf16* d, size_t n) {
        int n4 = (int)(n / 4);
        cvt_f32_bf16<<<dim3((n4 + 255) / 256), dim3(256), 0, stream>>>(s, d, n4);
    };
    cvt(x,    xb,   (size_t)BT * K_IN);
    cvt(fc1w, wfc1, (size_t)K_IN * K_IN);
    cvt(fc2w, wfc2, (size_t)DD * K_IN);
    cvt(unw,  wun,  (size_t)DD * DD);
    cvt(pxw,  wpx,  (size_t)DD * DD);
    cvt(pyw,  wpy,  (size_t)DD * DD);

    const dim3 blk(256);
    // fc1: h = relu(x @ fc1w^T + b)   [BT,1536]
    gemm_wmma<0><<<dim3(K_IN / 128, BT / 128), blk, 0, stream>>>(
        xb, wfc1, fc1b, BT, K_IN, K_IN, nullptr, hb, nullptr, nullptr);
    // fc2: u = h @ fc2w^T + b         [BT,768] (f32 + bf16)
    gemm_wmma<1><<<dim3(DD / 128, BT / 128), blk, 0, stream>>>(
        hb, wfc2, fc2b, BT, DD, K_IN, u, ubf, nullptr, nullptr);
    // unary: unp[m] += sum_n relu(u@unw^T + b)_n * unrw[n]
    gemm_wmma<2><<<dim3(DD / 128, BT / 128), blk, 0, stream>>>(
        ubf, wun, unb, BT, DD, DD, nullptr, nullptr, unrw, unp);
    // pairwise embeddings + row sumsq
    gemm_wmma<3><<<dim3(DD / 128, BT / 128), blk, 0, stream>>>(
        ubf, wpx, pxb, BT, DD, DD, xe, nullptr, nullptr, xss);
    gemm_wmma<3><<<dim3(DD / 128, BT / 128), blk, 0, stream>>>(
        ubf, wpy, pyb, BT, DD, DD, ye, nullptr, nullptr, yss);

    rinv_kernel<<<dim3((BT + 255) / 256), dim3(256), 0, stream>>>(xss, yss, xiv, yiv);
    ybar_kernel<<<dim3(DD / 256, BB), dim3(256), 0, stream>>>(ye, yiv, ybar);
    score_softmax_kernel<<<dim3(BB), dim3(256), 0, stream>>>(xe, ybar, xiv, unp, unrb, redw, wgt);
    out_kernel<<<dim3(DD / 256, BB), dim3(256), 0, stream>>>(u, wgt, out);
}
